// WMSA_86655260164340
// MI455X (gfx1250) — matrix operand order repaired and checked by reading, not compiled, not run on hardware
//
#include <hip/hip_runtime.h>
#include <hip/hip_bf16.h>
#include <stdint.h>

typedef __attribute__((ext_vector_type(8)))  _Float16 v8h;
typedef __attribute__((ext_vector_type(16))) _Float16 v16h;
typedef __attribute__((ext_vector_type(8)))  float    v8f;

#define NW   256
#define NP   256
#define CCH  192
#define NH   6
#define HD   32
#define NPIX (NW * NP)          // 65536

__device__ __forceinline__ v16h make16(v8h lo, v8h hi) {
  return __builtin_shufflevector(lo, hi, 0,1,2,3,4,5,6,7,8,9,10,11,12,13,14,15);
}

// A fragment (16xK slice, row-major, 32-wide K window starting at base).
// ISA layout: lane 0-15 -> M=lane, K = {kb..kb+7, 16+kb..16+kb+7}, kb = 0 (lo half) / 8 (hi half)
__device__ __forceinline__ v16h load_frag_a(const _Float16* base, int stride, int lane) {
  const int m  = lane & 15;
  const int kb = (lane & 16) ? 8 : 0;
  const _Float16* p = base + m * stride + kb;
  v8h lo = *(const v8h*)(p);
  v8h hi = *(const v8h*)(p + 16);
  return make16(lo, hi);
}

// B fragment (Kx16 slice). baseT is the TRANSPOSED operand: baseT[n][k], so each
// lane (N = lane&15) reads K = {kb..kb+7, 16+kb..16+kb+7}, kb = 0 / 8 per half-wave.
__device__ __forceinline__ v16h load_frag_b(const _Float16* baseT, int stride, int lane) {
  const int n  = lane & 15;
  const int kb = (lane & 16) ? 8 : 0;
  const _Float16* p = baseT + n * stride + kb;
  v8h lo = *(const v8h*)(p);
  v8h hi = *(const v8h*)(p + 16);
  return make16(lo, hi);
}

__device__ __forceinline__ v8f wmma_f16(v16h a, v16h b, v8f c) {
  return __builtin_amdgcn_wmma_f32_16x16x32_f16(false, a, false, b, (short)0, c, false, false);
}

// ---------------------------------------------------------------------------
// Kernel A: prep — x->f16, transpose+convert weights, precompute rpb gather.
// ---------------------------------------------------------------------------
__global__ void __launch_bounds__(256) prep_kernel(
    const float* __restrict__ x, const int* __restrict__ rpi,
    const float* __restrict__ qkv_w, const float* __restrict__ proj_w,
    const float* __restrict__ btab,
    _Float16* __restrict__ xh, _Float16* __restrict__ wT,
    _Float16* __restrict__ pT, float* __restrict__ rpb) {
  const long long N_X = (long long)NPIX * CCH;   // 12582912
  const long long N_W = 576 * CCH;               // 110592
  const long long N_P = CCH * CCH;               // 36864
  const long long N_R = (long long)NH * NP * NP; // 393216
  const long long total = N_X + N_W + N_P + N_R;
  const long long stride = (long long)gridDim.x * blockDim.x;
  for (long long i = (long long)blockIdx.x * blockDim.x + threadIdx.x; i < total; i += stride) {
    if (i < N_X) {
      xh[i] = (_Float16)x[i];
    } else if (i < N_X + N_W) {
      long long j = i - N_X;                     // wT[n][k] = qkv_w[k][n]
      int n = (int)(j / CCH), k = (int)(j % CCH);
      wT[j] = (_Float16)qkv_w[(long long)k * 576 + n];
    } else if (i < N_X + N_W + N_P) {
      long long j = i - N_X - N_W;               // pT[n][k] = proj_w[k][n]
      int n = (int)(j / CCH), k = (int)(j % CCH);
      pT[j] = (_Float16)proj_w[(long long)k * CCH + n];
    } else {
      long long j = i - N_X - N_W - N_P;         // rpb[h][i][j] = btab[rpi[i][j]][h]
      int h  = (int)(j >> 16);
      int ij = (int)(j & 0xFFFF);
      rpb[j] = btab[(long long)rpi[ij] * NH + h];
    }
  }
}

// ---------------------------------------------------------------------------
// Kernel B: QKV GEMM (65536x192) @ (192x576) -> scatter Q(scaled)/K/V as f16
// tiles laid out (w, head, pixel, d) so attention fragments are contiguous.
// Block = 8 waves covering 32(M) x 64(N); wave = one 16x16 tile, 6 K-chunks.
// ---------------------------------------------------------------------------
__global__ void __launch_bounds__(256) qkv_gemm(
    const _Float16* __restrict__ xh, const _Float16* __restrict__ wT,
    const float* __restrict__ qkv_b,
    _Float16* __restrict__ Qh, _Float16* __restrict__ Kh, _Float16* __restrict__ Vh) {
  const int lane = threadIdx.x & 31;
  const int wv   = threadIdx.x >> 5;
  const int m0 = (blockIdx.x * 2 + (wv & 1)) * 16;
  const int n0 = (blockIdx.y * 4 + (wv >> 1)) * 16;
  const _Float16* arow = xh + (long long)m0 * CCH;
  const _Float16* brow = wT + (long long)n0 * CCH;
  v8f acc = {};
#pragma unroll
  for (int kk = 0; kk < CCH; kk += 32) {
    v16h a = load_frag_a(arow + kk, CCH, lane);
    v16h b = load_frag_b(brow + kk, CCH, lane);
    acc = wmma_f16(a, b, acc);
  }
  // Epilogue: n in [0,576): 0..191 -> Q, 192..383 -> K, 384..575 -> V
  const int n    = n0 + (lane & 15);
  const int qkvi = n / CCH;
  const int rem  = n % CCH;
  const int head = rem >> 5;
  const int dd   = rem & 31;
  const float bias = qkv_b[n];
  const float scl  = (qkvi == 0) ? 0.17677669529663687f : 1.0f;  // d^-0.5 on Q
  _Float16* dst = (qkvi == 0) ? Qh : ((qkvi == 1) ? Kh : Vh);
  const int hi8 = (lane & 16) ? 8 : 0;
#pragma unroll
  for (int r = 0; r < 8; r++) {
    int m = m0 + r + hi8;
    int w = m >> 8, p = m & 255;
    dst[((long long)(w * NH + head) * NP + p) * HD + dd] = (_Float16)((acc[r] + bias) * scl);
  }
}

// ---------------------------------------------------------------------------
// Kernel C: fused attention, one block per (window, head). K staged into LDS
// with gfx1250 async-to-LDS b128 DMA (ASYNCcnt); V^T staged with conflict-free
// contiguous LDS writes. S held in registers (16 rows x 256 cols per wave);
// softmax via half-wave shfl_xor; P re-laid out C->A via 1KB private LDS slice.
// ---------------------------------------------------------------------------
__global__ void __launch_bounds__(256) attn_kernel(
    const _Float16* __restrict__ Qh, const _Float16* __restrict__ Kh,
    const _Float16* __restrict__ Vh, const float* __restrict__ rpb,
    const float* __restrict__ mask, _Float16* __restrict__ ah) {
  __shared__ _Float16 Kl[NP * HD];       // 16 KB, (pixel, d)
  __shared__ _Float16 Vt[HD * NP];       // 16 KB, (d, pixel)  == V transposed
  __shared__ _Float16 Ps[8][16 * 32];    //  8 KB, per-wave P chunk staging
  const int w = blockIdx.x, h = blockIdx.y;
  const int tid = threadIdx.x, lane = tid & 31, wv = tid >> 5;
  const long long base = (long long)(w * NH + h) * NP * HD;   // 8192 halfs

  // --- Stage K: async global->LDS DMA, 16 B per lane per iter (no VGPR data).
  {
    const unsigned ldsK = (unsigned)(uintptr_t)(&Kl[0]);
    const unsigned long long gk = (unsigned long long)(const void*)(Kh + base);
    // 1024 x 16B = 16 KB; 256 threads -> 4 iters each. GVS mode: s[base]+v[off].
#pragma unroll
    for (int it = 0; it < 4; it++) {
      int i = tid + it * 256;
      unsigned loff = ldsK + (unsigned)i * 16u;
      unsigned voff = (unsigned)i * 16u;
      asm volatile("global_load_async_to_lds_b128 %0, %1, %2"
                   :: "v"(loff), "v"(voff), "s"(gk) : "memory");
    }
  }
  // --- Stage V transposed: i -> (dd=i>>8, p=i&255) so LDS writes are
  //     contiguous (conflict-free); strided global reads hit L0/L2.
  {
    const _Float16* vsrc = Vh + base;
    for (int i = tid; i < NP * HD; i += 256) {
      int dd = i >> 8, p = i & 255;
      Vt[i] = vsrc[p * HD + dd];
    }
  }
  asm volatile("s_wait_asynccnt 0x0" ::: "memory");
  __syncthreads();

  const int hi8 = (lane & 16) ? 8 : 0;
  for (int pass = 0; pass < 2; pass++) {
    const int rowBase = (pass * 8 + wv) * 16;
    // Q fragment: full K-dim (d=32) in one x32 fragment.
    v16h qa = load_frag_a(Qh + base + (long long)rowBase * HD, HD, lane);

    // S = Q @ K^T : 16 tiles of 16x16, one WMMA each.
    v8f S[16];
#pragma unroll
    for (int t = 0; t < 16; t++) {
      v16h kb = load_frag_b(Kl + t * 16 * HD, HD, lane);
      v8f z = {};
      S[t] = wmma_f16(qa, kb, z);
    }

    // + relative position bias + attention mask
    const float* rp = rpb + ((long long)h * NP + rowBase) * NP;
    const float* mk = mask + ((long long)w * NP + rowBase) * NP;
#pragma unroll
    for (int t = 0; t < 16; t++) {
      int col = t * 16 + (lane & 15);
#pragma unroll
      for (int r = 0; r < 8; r++) {
        int row = r + hi8;
        S[t][r] += rp[row * NP + col] + mk[row * NP + col];
      }
    }

    // Row softmax: N = lane within 16-lane half, so reduce with xor 1,2,4,8.
#pragma unroll
    for (int r = 0; r < 8; r++) {
      float m = -3.0e38f;
#pragma unroll
      for (int t = 0; t < 16; t++) m = fmaxf(m, S[t][r]);
      m = fmaxf(m, __shfl_xor(m, 1, 32));
      m = fmaxf(m, __shfl_xor(m, 2, 32));
      m = fmaxf(m, __shfl_xor(m, 4, 32));
      m = fmaxf(m, __shfl_xor(m, 8, 32));
      float s = 0.0f;
#pragma unroll
      for (int t = 0; t < 16; t++) { float e = __expf(S[t][r] - m); S[t][r] = e; s += e; }
      s += __shfl_xor(s, 1, 32);
      s += __shfl_xor(s, 2, 32);
      s += __shfl_xor(s, 4, 32);
      s += __shfl_xor(s, 8, 32);
      float inv = 1.0f / s;
#pragma unroll
      for (int t = 0; t < 16; t++) S[t][r] *= inv;
    }

    // O = P @ V : 8 K-chunks of 32; restage P chunk (C-layout -> A-layout) via LDS.
    v8f O0 = {}, O1 = {};
    _Float16* slice = &Ps[wv][0];
    for (int c = 0; c < 8; c++) {
#pragma unroll
      for (int t = 0; t < 2; t++) {
        int col = t * 16 + (lane & 15);
#pragma unroll
        for (int r = 0; r < 8; r++)
          slice[(r + hi8) * 32 + col] = (_Float16)S[2 * c + t][r];
      }
      v16h pa  = load_frag_a(slice, 32, lane);
      v16h vb0 = load_frag_b(Vt + c * 32, NP, lane);           // d 0..15
      v16h vb1 = load_frag_b(Vt + 16 * NP + c * 32, NP, lane); // d 16..31
      O0 = wmma_f16(pa, vb0, O0);
      O1 = wmma_f16(pa, vb1, O1);
    }

    // Write (w, pixel, h*32+dd) as f16 for the projection GEMM.
#pragma unroll
    for (int r = 0; r < 8; r++) {
      int row = rowBase + r + hi8;
      long long o = ((long long)w * NP + row) * CCH + h * HD;
      ah[o + (lane & 15)]      = (_Float16)O0[r];
      ah[o + 16 + (lane & 15)] = (_Float16)O1[r];
    }
  }
}

// ---------------------------------------------------------------------------
// Kernel D: projection GEMM (65536x192) @ (192x192) + bias -> f32 output.
// ---------------------------------------------------------------------------
__global__ void __launch_bounds__(256) proj_gemm(
    const _Float16* __restrict__ ah, const _Float16* __restrict__ pT,
    const float* __restrict__ proj_b, float* __restrict__ out) {
  const int lane = threadIdx.x & 31;
  const int wv   = threadIdx.x >> 5;
  const int m0 = (blockIdx.x * 2 + (wv & 1)) * 16;
  const int n0 = (blockIdx.y * 4 + (wv >> 1)) * 16;
  const _Float16* arow = ah + (long long)m0 * CCH;
  const _Float16* brow = pT + (long long)n0 * CCH;
  v8f acc = {};
#pragma unroll
  for (int kk = 0; kk < CCH; kk += 32) {
    v16h a = load_frag_a(arow + kk, CCH, lane);
    v16h b = load_frag_b(brow + kk, CCH, lane);
    acc = wmma_f16(a, b, acc);
  }
  const int n = n0 + (lane & 15);
  const float bias = proj_b[n];
  const int hi8 = (lane & 16) ? 8 : 0;
#pragma unroll
  for (int r = 0; r < 8; r++) {
    int m = m0 + r + hi8;
    out[(long long)m * CCH + n] = acc[r] + bias;
  }
}

// ---------------------------------------------------------------------------
extern "C" void kernel_launch(void* const* d_in, const int* in_sizes, int n_in,
                              void* d_out, int out_size, void* d_ws, size_t ws_size,
                              hipStream_t stream) {
  const float* x      = (const float*)d_in[0];
  const int*   rpi    = (const int*)d_in[1];
  const float* mask   = (const float*)d_in[2];
  const float* qkv_w  = (const float*)d_in[3];
  const float* qkv_b  = (const float*)d_in[4];
  const float* proj_w = (const float*)d_in[5];
  const float* proj_b = (const float*)d_in[6];
  const float* btab   = (const float*)d_in[7];
  float* out = (float*)d_out;

  char* ws = (char*)d_ws;
  size_t off = 0;
  _Float16* xh = (_Float16*)(ws + off); off += (size_t)NPIX * CCH * 2;   // reused as ah
  _Float16* wT = (_Float16*)(ws + off); off += (size_t)576 * CCH * 2;
  _Float16* pT = (_Float16*)(ws + off); off += (size_t)CCH * CCH * 2;
  float*   rpb = (float*)(ws + off);    off += (size_t)NH * NP * NP * 4;
  const size_t qkv_elems = (size_t)NW * NH * NP * HD;    // 12,582,912 halfs each
  _Float16* Qh = (_Float16*)(ws + off); off += qkv_elems * 2;
  _Float16* Kh = (_Float16*)(ws + off); off += qkv_elems * 2;
  _Float16* Vh = (_Float16*)(ws + off); off += qkv_elems * 2;

  prep_kernel<<<dim3(6144), dim3(256), 0, stream>>>(x, rpi, qkv_w, proj_w, btab,
                                                    xh, wT, pT, rpb);
  qkv_gemm<<<dim3(2048, 9), dim3(256), 0, stream>>>(xh, wT, qkv_b, Qh, Kh, Vh);
  attn_kernel<<<dim3(NW, NH), dim3(256), 0, stream>>>(Qh, Kh, Vh, rpb, mask, xh /*ah*/);
  proj_gemm<<<dim3(2048, 3), dim3(256), 0, stream>>>(xh /*ah*/, pT, proj_b, out);
}
